// MoELayer_34892314313339
// MI455X (gfx1250) — matrix operand order repaired and checked by reading, not compiled, not run on hardware
//
#include <hip/hip_runtime.h>
#include <hip/hip_bf16.h>
#include <math.h>

// Problem constants (from reference): N=8192 tokens, D=2048, H=2048, E=8 experts, K=2
#define NTOK 8192
#define DDIM 2048
#define HDIM 2048
#define NEXP 8
#define KSEL 2

// Block GEMM tile: 32 (M) x 256 (N) x 32 (K). 8 waves: 2 wave-rows x 4 wave-cols,
// each wave owns a 16x64 slab = 4 x v_wmma_f32_16x16x32_bf16 accumulators.
#define TM 32
#define TN 256
#define TK 32
#define LDSPAD 8                 // pad rows to 40 bf16 (80 B): bank step 20 -> no conflicts
#define TKP (TK + LDSPAD)

typedef __attribute__((ext_vector_type(16))) __bf16 v16bf;
typedef __attribute__((ext_vector_type(8)))  __bf16 v8bf;
typedef __attribute__((ext_vector_type(4)))  __bf16 v4bf;
typedef __attribute__((ext_vector_type(8)))  float  v8f;

#define SHUF16(a, b) __builtin_shufflevector(a, b, 0,1,2,3,4,5,6,7,8,9,10,11,12,13,14,15)

// ---------------------------------------------------------------------------
// Workspace layout (bytes):
//   0        : topi     int  [N*K]      (65536 B)
//   65536    : topw     float[N*K]      (65536 B)
//   131072   : counters int  [64]       (counts[0..7], counts2[8..15], offsets[16..24])
//   132096   : slot_tok int  [N*K]      (65536 B)
//   197632   : slot_w   float[N*K]      (65536 B)
//   263168   : h_ws     bf16 [N*K*H]    (64 MB)  -- layer-1 activations per routed slot
// ---------------------------------------------------------------------------

__global__ void zero_kernel(float* __restrict__ out, size_t n, int* __restrict__ cnt) {
    size_t i = (size_t)blockIdx.x * blockDim.x + threadIdx.x;
    size_t stride = (size_t)gridDim.x * blockDim.x;
    for (size_t j = i; j < n; j += stride) out[j] = 0.f;
    if (i < 64) cnt[i] = 0;
}

// One wave per token: 8-expert logits, cross-lane reduce, top-2 + softmax(2).
__global__ __launch_bounds__(256)
void gate_kernel(const float* __restrict__ x, const float* __restrict__ Wg,
                 const float* __restrict__ bg, int* __restrict__ topi,
                 float* __restrict__ topw) {
    const int lane = threadIdx.x & 31;
    const int n = blockIdx.x * (blockDim.x >> 5) + (threadIdx.x >> 5);
    if (n >= NTOK) return;
    float acc[NEXP];
#pragma unroll
    for (int e = 0; e < NEXP; ++e) acc[e] = 0.f;
    const float* xr = x + (size_t)n * DDIM;
    for (int d = lane; d < DDIM; d += 32) {
        const float xv = xr[d];
        const float* wr = Wg + (size_t)d * NEXP;   // Wg is [D,E] row-major, 8 contiguous floats
#pragma unroll
        for (int e = 0; e < NEXP; ++e) acc[e] += xv * wr[e];
    }
#pragma unroll
    for (int off = 16; off >= 1; off >>= 1) {
#pragma unroll
        for (int e = 0; e < NEXP; ++e) acc[e] += __shfl_xor(acc[e], off, 32);
    }
    if (lane == 0) {
#pragma unroll
        for (int e = 0; e < NEXP; ++e) acc[e] += bg[e];
        int i0 = 0;
#pragma unroll
        for (int e = 1; e < NEXP; ++e) if (acc[e] > acc[i0]) i0 = e;
        int i1 = (i0 == 0) ? 1 : 0;
#pragma unroll
        for (int e = 0; e < NEXP; ++e) if (e != i0 && acc[e] > acc[i1]) i1 = e;
        const float z = __expf(acc[i1] - acc[i0]);   // softmax over the 2 selected logits
        const float s = 1.f / (1.f + z);
        topi[n * 2 + 0] = i0; topi[n * 2 + 1] = i1;
        topw[n * 2 + 0] = s;  topw[n * 2 + 1] = z * s;
    }
}

__global__ void count_kernel(const int* __restrict__ topi, int* __restrict__ counts) {
    const int n = blockIdx.x * blockDim.x + threadIdx.x;
    if (n >= NTOK) return;
    atomicAdd(&counts[topi[n * 2 + 0]], 1);
    atomicAdd(&counts[topi[n * 2 + 1]], 1);
}

__global__ void offsets_kernel(const int* __restrict__ counts, int* __restrict__ offsets) {
    if (threadIdx.x == 0) {
        int acc = 0;
        for (int e = 0; e < NEXP; ++e) { offsets[e] = acc; acc += counts[e]; }
        offsets[NEXP] = acc;
    }
}

__global__ void scatter_kernel(const int* __restrict__ topi, const float* __restrict__ topw,
                               const int* __restrict__ offsets, int* __restrict__ counts2,
                               int* __restrict__ slot_tok, float* __restrict__ slot_w) {
    const int n = blockIdx.x * blockDim.x + threadIdx.x;
    if (n >= NTOK) return;
#pragma unroll
    for (int k = 0; k < KSEL; ++k) {
        const int e = topi[n * 2 + k];
        const int p = offsets[e] + atomicAdd(&counts2[e], 1);
        slot_tok[p] = n;
        slot_w[p]   = topw[n * 2 + k];
    }
}

// Stage one 32(K) x 256(N) fp32 weight tile into LDS as bf16, K-contiguous per
// column (transposed), so WMMA B fragments are wide ds_load_b128 reads.
// Thread map: tid&31 -> 8-column group (coalesced float4 global reads),
//             tid>>5 -> 4-row K group (packed v4bf LDS stores).
__device__ __forceinline__ void stage_B(const float* __restrict__ Wrow0, int tid,
                                        __bf16 (*Bs)[TKP]) {
    const int colbase = (tid & 31) * 8;
    const int kr = (tid >> 5) * 4;
    float rows[4][8];
#pragma unroll
    for (int r = 0; r < 4; ++r) {
        const float4 u0 = *(const float4*)(Wrow0 + (size_t)(kr + r) * HDIM + colbase);
        const float4 u1 = *(const float4*)(Wrow0 + (size_t)(kr + r) * HDIM + colbase + 4);
        rows[r][0] = u0.x; rows[r][1] = u0.y; rows[r][2] = u0.z; rows[r][3] = u0.w;
        rows[r][4] = u1.x; rows[r][5] = u1.y; rows[r][6] = u1.z; rows[r][7] = u1.w;
    }
#pragma unroll
    for (int c = 0; c < 8; ++c) {
        v4bf p = { (__bf16)rows[0][c], (__bf16)rows[1][c],
                   (__bf16)rows[2][c], (__bf16)rows[3][c] };
        *(v4bf*)&Bs[colbase + c][kr] = p;
    }
}

// ---------------------------------------------------------------------------
// Layer 1: h[slot, :] = relu(x[token(slot)] @ W1[e] + b1[e])  -> bf16 workspace
// ---------------------------------------------------------------------------
__global__ __launch_bounds__(256)
void ffn1_kernel(const float* __restrict__ x, const float* __restrict__ W1,
                 const float* __restrict__ b1, const int* __restrict__ offsets,
                 const int* __restrict__ slot_tok, __bf16* __restrict__ h_ws) {
    const int e = blockIdx.z;
    const int seg_base = offsets[e];
    const int seg_end  = offsets[e + 1];
    const int cnt = seg_end - seg_base;
    const int mtile = blockIdx.x;
    if (mtile * TM >= cnt) return;                  // block-uniform early exit
    const int nb = blockIdx.y * TN;

    __shared__ __bf16 As[TM][TKP];                  // gathered x rows (K-contiguous)
    __shared__ __bf16 Bs[TN][TKP];                  // W1 tile, transposed (K-contiguous)
    __shared__ int rowTok[TM];

    const int tid  = threadIdx.x;
    const int lane = tid & 31;
    const int wave = tid >> 5;
    const int wm   = wave >> 2;                     // 0..1
    const int wn   = wave & 3;                      // 0..3

    if (tid < TM) {
        const int s = seg_base + mtile * TM + tid;
        rowTok[tid] = (s < seg_end) ? slot_tok[s] : slot_tok[seg_base];
    }
    __syncthreads();

    const float* Wbase = W1 + (size_t)e * DDIM * HDIM;

    v8f acc[4] = {};

    const int ar = tid >> 3;                        // A stage: row 0..31
    const int ak = (tid & 7) * 4;                   //          4 K values

    // WMMA fragment coordinates (ISA wave32 layouts)
    const int mrow = wm * 16 + (lane & 15);         // A: lane&15 = M row
    const int kh   = (lane >> 4) * 8;               // A: K chunks {0-7,16-23} / {8-15,24-31}
    const int kb   = (lane >> 4) * 16;              // unused; B reads take the same kh chunks as A
    const int nc   = lane & 15;                     // B/C: lane&15 = N column

    for (int k0 = 0; k0 < DDIM; k0 += TK) {
        {   // stage A (32x32), packed v4bf stores
            const float4 v = *(const float4*)(x + (size_t)rowTok[ar] * DDIM + k0 + ak);
            v4bf p = { (__bf16)v.x, (__bf16)v.y, (__bf16)v.z, (__bf16)v.w };
            *(v4bf*)&As[ar][ak] = p;
        }
        stage_B(Wbase + (size_t)k0 * HDIM + nb, tid, Bs);
        if (k0 + TK < DDIM)                          // global_prefetch_b8 on next W tile
            __builtin_prefetch(Wbase + (size_t)(k0 + TK + (tid >> 5) * 4) * HDIM + nb +
                               (tid & 31) * 8, 0, 1);
        __syncthreads();

        const v8bf a0 = *(const v8bf*)&As[mrow][kh];
        const v8bf a1 = *(const v8bf*)&As[mrow][16 + kh];
        const v16bf a = SHUF16(a0, a1);
#pragma unroll
        for (int j = 0; j < 4; ++j) {
            const __bf16* bp = &Bs[wn * 64 + j * 16 + nc][kh];
            const v8bf b0 = *(const v8bf*)(bp);
            const v8bf b1 = *(const v8bf*)(bp + 16);
            const v16bf b = SHUF16(b0, b1);
            acc[j] = __builtin_amdgcn_wmma_f32_16x16x32_bf16(
                false, a, false, b, (short)0, acc[j], false, false);
        }
        __syncthreads();
    }

    // Epilogue: bias + ReLU, store bf16 activations per routed slot.
#pragma unroll
    for (int j = 0; j < 4; ++j) {
        const int col = nb + wn * 64 + j * 16 + nc;
        const float bias = b1[(size_t)e * HDIM + col];
#pragma unroll
        for (int r = 0; r < 8; ++r) {
            const int row = mtile * TM + wm * 16 + (lane >> 4) * 8 + r;
            if (row < cnt) {
                float v = acc[j][r] + bias;
                v = v > 0.f ? v : 0.f;
                h_ws[(size_t)(seg_base + row) * HDIM + col] = (__bf16)v;
            }
        }
    }
}

// ---------------------------------------------------------------------------
// Layer 2: out[token] += gate_w * (h[slot] @ W2[e] + b2[e])   (2 adds/token -> exact)
// ---------------------------------------------------------------------------
__global__ __launch_bounds__(256)
void ffn2_kernel(const __bf16* __restrict__ h_ws, const float* __restrict__ W2,
                 const float* __restrict__ b2, const int* __restrict__ offsets,
                 const int* __restrict__ slot_tok, const float* __restrict__ slot_w,
                 float* __restrict__ out) {
    const int e = blockIdx.z;
    const int seg_base = offsets[e];
    const int seg_end  = offsets[e + 1];
    const int cnt = seg_end - seg_base;
    const int mtile = blockIdx.x;
    if (mtile * TM >= cnt) return;
    const int nb = blockIdx.y * TN;

    __shared__ __bf16 Bs[TN][TKP];

    const int tid  = threadIdx.x;
    const int lane = tid & 31;
    const int wave = tid >> 5;
    const int wm   = wave >> 2;
    const int wn   = wave & 3;

    v8f acc[4] = {};

    // A rows come straight from the bf16 activation workspace (already bf16,
    // K-contiguous in memory -> two 16B global loads per lane per K-step).
    int arow = mtile * TM + wm * 16 + (lane & 15);
    if (arow >= cnt) arow = 0;                       // clamp; masked at store
    const __bf16* ha = h_ws + (size_t)(seg_base + arow) * HDIM;
    const int kh = (lane >> 4) * 8;
    const int kb = (lane >> 4) * 16;
    const int nc = lane & 15;

    const float* Wbase = W2 + (size_t)e * HDIM * HDIM;

    for (int k0 = 0; k0 < HDIM; k0 += TK) {
        stage_B(Wbase + (size_t)k0 * HDIM + nb, tid, Bs);
        if (k0 + TK < HDIM)
            __builtin_prefetch(Wbase + (size_t)(k0 + TK + (tid >> 5) * 4) * HDIM + nb +
                               (tid & 31) * 8, 0, 1);
        __syncthreads();

        const v8bf a0 = *(const v8bf*)(ha + k0 + kh);
        const v8bf a1 = *(const v8bf*)(ha + k0 + 16 + kh);
        const v16bf a = SHUF16(a0, a1);
#pragma unroll
        for (int j = 0; j < 4; ++j) {
            const __bf16* bp = &Bs[wn * 64 + j * 16 + nc][kh];
            const v8bf b0 = *(const v8bf*)(bp);
            const v8bf b1 = *(const v8bf*)(bp + 16);
            const v16bf b = SHUF16(b0, b1);
            acc[j] = __builtin_amdgcn_wmma_f32_16x16x32_bf16(
                false, a, false, b, (short)0, acc[j], false, false);
        }
        __syncthreads();
    }

#pragma unroll
    for (int j = 0; j < 4; ++j) {
        const int col = nb + wn * 64 + j * 16 + nc;
        const float bias = b2[(size_t)e * HDIM + col];
#pragma unroll
        for (int r = 0; r < 8; ++r) {
            const int row = mtile * TM + wm * 16 + (lane >> 4) * 8 + r;
            if (row < cnt) {
                const int s = seg_base + row;
                const float w = slot_w[s];
                const int tok = slot_tok[s];
                atomicAdd(&out[(size_t)tok * HDIM + col], w * (acc[j][r] + bias));
            }
        }
    }
}

// ---------------------------------------------------------------------------
extern "C" void kernel_launch(void* const* d_in, const int* in_sizes, int n_in,
                              void* d_out, int out_size, void* d_ws, size_t ws_size,
                              hipStream_t stream) {
    const float* x  = (const float*)d_in[0];
    const float* Wg = (const float*)d_in[1];
    const float* bg = (const float*)d_in[2];
    const float* W1 = (const float*)d_in[3];
    const float* b1 = (const float*)d_in[4];
    const float* W2 = (const float*)d_in[5];
    const float* b2 = (const float*)d_in[6];
    float* out = (float*)d_out;

    char* ws = (char*)d_ws;
    int*    topi     = (int*)(ws + 0);
    float*  topw     = (float*)(ws + 65536);
    int*    counters = (int*)(ws + 131072);     // counts[0..7], counts2[8..15], offsets[16..24]
    int*    counts   = counters;
    int*    counts2  = counters + 8;
    int*    offsets  = counters + 16;
    int*    slot_tok = (int*)(ws + 132096);
    float*  slot_w   = (float*)(ws + 197632);
    __bf16* h_ws     = (__bf16*)(ws + 263168);

    zero_kernel<<<4096, 256, 0, stream>>>(out, (size_t)NTOK * HDIM, counters);
    gate_kernel<<<NTOK / 8, 256, 0, stream>>>(x, Wg, bg, topi, topw);
    count_kernel<<<NTOK / 256, 256, 0, stream>>>(topi, counts);
    offsets_kernel<<<1, 32, 0, stream>>>(counts, offsets);
    scatter_kernel<<<NTOK / 256, 256, 0, stream>>>(topi, topw, offsets, counts2,
                                                   slot_tok, slot_w);

    dim3 gemm_grid(NTOK / TM, HDIM / TN, NEXP);   // worst-case m-tiles; blocks early-exit
    ffn1_kernel<<<gemm_grid, 256, 0, stream>>>(x, W1, b1, offsets, slot_tok, h_ws);
    ffn2_kernel<<<gemm_grid, 256, 0, stream>>>(h_ws, W2, b2, offsets, slot_tok, slot_w, out);
}
